// GCNPose3D_15006615732993
// MI455X (gfx1250) — matrix operand order repaired and checked by reading, not compiled, not run on hardware
//
#include <hip/hip_runtime.h>
#include <math.h>

typedef __attribute__((ext_vector_type(16))) _Float16 v16h;
typedef __attribute__((ext_vector_type(8)))  _Float16 v8h;
typedef __attribute__((ext_vector_type(4)))  _Float16 v4h;
typedef __attribute__((ext_vector_type(8)))  float    v8f;
typedef __attribute__((ext_vector_type(4)))  float    v4f;

#define NJ      17
#define H       256
#define NB      3
#define ROWS    (NB * NJ)     // 51 real rows per tile
#define MT      64            // padded M tile (4 x 16)
#define SH      264           // half-element row stride for h_f16 (pad to dodge bank conflicts)
#define SF      264           // float-element row stride
#define KBLK    8             // 256 / 32 k-steps
#define LAYERS  4

// ---------------------------------------------------------------------------
// Pre-kernel: convert Ws (4,256,256) fp32 -> f16 packed in WMMA B-fragment
// order.  For V_WMMA_F32_16X16X32_F16 this code lays out the B matrix (K=32 x N=16)
// as: lane = n + 16*(k/16 within the 32-block), halves 0..15 of the lane = K
// 0..15 of that half-block.  We store each (kb, nt) fragment as 32 lanes x 16
// contiguous halves so the GEMM kernel does one aligned 32B load per fragment.
// ---------------------------------------------------------------------------
__global__ void pack_w_f16(const float* __restrict__ Ws, _Float16* __restrict__ wq) {
    int i = blockIdx.x * 256 + threadIdx.x;
    if (i >= LAYERS * H * H) return;
    int l   = i >> 16;          // / 65536
    int rem = i & 65535;
    int k   = rem >> 8;         // row of W (input channel)
    int n   = rem & 255;        // col of W (output channel)
    int kb    = k >> 5;
    int kin   = k & 31;
    int khalf = kin >> 4;       // which 16-K half-block
    int hh    = kin & 15;       // half index within lane
    int nt    = n >> 4;
    int nl    = n & 15;
    int lane  = nl + (khalf << 4);
    size_t dst = ((size_t)((l * KBLK + kb) * 16 + nt)) * 512 + (size_t)lane * 16 + hh;
    wq[dst] = (_Float16)Ws[i];
}

// ---------------------------------------------------------------------------
// Fused GCN: input proj -> 4x (h@W via WMMA, A_hat mix + relu + residual)
// -> output proj.  One workgroup = 3 batch entries (51 rows, padded to 64),
// 256 threads = 8 waves; wave w owns N-tiles {2w, 2w+1} x all 4 M-tiles.
// ---------------------------------------------------------------------------
__global__ __launch_bounds__(256) void gcn_fused(
    const float* __restrict__ x, const float* __restrict__ adj,
    const float* __restrict__ w_in, const float* __restrict__ b_in,
    const _Float16* __restrict__ wq, const float* __restrict__ bs,
    const float* __restrict__ w_out, const float* __restrict__ b_out,
    float* __restrict__ y, int BT)
{
    __shared__ _Float16 sh_h16[MT * SH];   // f16 copy of h (WMMA A operand)
    __shared__ float    sh_h32[MT * SF];   // fp32 h (residual path)
    __shared__ float    sh_sup[MT * SF];   // fp32 support = h @ W
    __shared__ float    sh_A[NJ * NJ];     // normalized adjacency
    __shared__ float    sh_dh[NJ];

    const int tid  = threadIdx.x;
    const int lane = tid & 31;
    const int wave = tid >> 5;
    const int bt0  = blockIdx.x * NB;

    int vrows = (BT - bt0) * NJ;           // guard for a ragged last tile
    if (vrows > ROWS) vrows = ROWS;

    // ---- A_hat = D^-1/2 A D^-1/2 -------------------------------------------
    if (tid < NJ) {
        float s = 0.f;
        for (int k = 0; k < NJ; ++k) s += adj[tid * NJ + k];
        sh_dh[tid] = 1.0f / sqrtf(s + 1e-5f);
    }
    __syncthreads();
    for (int t = tid; t < NJ * NJ; t += 256) {
        int j = t / NJ, k = t - j * NJ;
        sh_A[t] = sh_dh[j] * adj[t] * sh_dh[k];
    }

    // ---- input projection: h0 = x @ w_in + b_in ----------------------------
    const float* xg = x + (size_t)bt0 * (NJ * 3);
    for (int idx = tid; idx < MT * H; idx += 256) {
        int row = idx >> 8, d = idx & 255;
        float v = 0.f;
        if (row < vrows) {
            float x0 = xg[row * 3 + 0], x1 = xg[row * 3 + 1], x2 = xg[row * 3 + 2];
            v = b_in[d] + x0 * w_in[d] + x1 * w_in[H + d] + x2 * w_in[2 * H + d];
            sh_h32[row * SF + d] = v;
        }
        sh_h16[row * SH + d] = (_Float16)v;   // pad rows contribute zeros
    }

    const int nt0 = wave * 2;
    const int mlo = lane & 15;     // M within tile / N within tile
    const int sel = lane >> 4;     // K-half select for A, row-half for D

    for (int l = 0; l < LAYERS; ++l) {
        __syncthreads();           // h16 from previous phase visible

        if (tid == 0 && l + 1 < LAYERS)
            __builtin_prefetch(wq + (size_t)(l + 1) * (H * H), 0, 1);

        // ---- support = h @ W[l] via WMMA -----------------------------------
        v8f zero = {};
        v8f acc[4][2];
#pragma unroll
        for (int mt = 0; mt < 4; ++mt)
#pragma unroll
            for (int tn = 0; tn < 2; ++tn) acc[mt][tn] = zero;

        const _Float16* wl = wq + (size_t)l * (H * H);
#pragma unroll
        for (int kb = 0; kb < KBLK; ++kb) {
            v16h afrag[4];
#pragma unroll
            for (int mt = 0; mt < 4; ++mt) {
                // A layout (matches packed B): lane holds row (lane%16); halves 0..15 =
                // K sel*16..sel*16+15 of this k-block
                const _Float16* ap = sh_h16 + (mt * 16 + mlo) * SH + kb * 32 + sel * 16;
                v8h lo = *(const v8h*)ap;
                v8h hi = *(const v8h*)(ap + 8);
                afrag[mt] = __builtin_shufflevector(lo, hi,
                    0, 1, 2, 3, 4, 5, 6, 7, 8, 9, 10, 11, 12, 13, 14, 15);
            }
#pragma unroll
            for (int tn = 0; tn < 2; ++tn) {
                const v16h bfrag =
                    *(const v16h*)(wl + ((size_t)(kb * 16 + nt0 + tn)) * 512 + (size_t)lane * 16);
#pragma unroll
                for (int mt = 0; mt < 4; ++mt) {
                    acc[mt][tn] = __builtin_amdgcn_wmma_f32_16x16x32_f16(
                        false, afrag[mt], false, bfrag, (short)0, acc[mt][tn], false, false);
                }
            }
        }

        // ---- spill accumulators to LDS (D layout: n=lane%16, m=r+8*sel) ----
#pragma unroll
        for (int mt = 0; mt < 4; ++mt)
#pragma unroll
            for (int tn = 0; tn < 2; ++tn) {
                int n     = (nt0 + tn) * 16 + mlo;
                int rbase = mt * 16 + sel * 8;
#pragma unroll
                for (int r = 0; r < 8; ++r)
                    sh_sup[(rbase + r) * SF + n] = acc[mt][tn][r];
            }
        __syncthreads();

        // ---- out = A_hat @ support + b ; h = relu(out) + h -----------------
        int row = tid & 63;
        int q   = tid >> 6;                 // feature quarter (64 features)
        if (row < vrows) {
            int b = row / NJ;
            int j = row - b * NJ;
            const float* Aj  = sh_A + j * NJ;
            const float* bsl = bs + l * H;
            int d0 = q * 64;
            for (int d = d0; d < d0 + 64; d += 4) {
                v4f a4 = { bsl[d], bsl[d + 1], bsl[d + 2], bsl[d + 3] };
#pragma unroll
                for (int k = 0; k < NJ; ++k) {
                    float a = Aj[k];
                    const v4f s4 = *(const v4f*)&sh_sup[(b * NJ + k) * SF + d];
                    a4 += a * s4;
                }
                v4f h4 = *(const v4f*)&sh_h32[row * SF + d];
                v4f r4;
#pragma unroll
                for (int c = 0; c < 4; ++c) r4[c] = fmaxf(a4[c], 0.f) + h4[c];
                *(v4f*)&sh_h32[row * SF + d] = r4;
                v4h r16 = { (_Float16)r4[0], (_Float16)r4[1], (_Float16)r4[2], (_Float16)r4[3] };
                *(v4h*)&sh_h16[row * SH + d] = r16;
            }
        }
    }
    __syncthreads();

    // ---- output projection: y = h @ w_out + b_out --------------------------
    for (int t = tid; t < vrows * 3; t += 256) {
        int row = t / 3, oc = t - row * 3;
        float a = b_out[oc];
        for (int d = 0; d < H; ++d) a += sh_h32[row * SF + d] * w_out[d * 3 + oc];
        y[((size_t)bt0 * NJ + row) * 3 + oc] = a;
    }
}

// ---------------------------------------------------------------------------
extern "C" void kernel_launch(void* const* d_in, const int* in_sizes, int n_in,
                              void* d_out, int out_size, void* d_ws, size_t ws_size,
                              hipStream_t stream) {
    const float* x     = (const float*)d_in[0];
    const float* adj   = (const float*)d_in[1];
    const float* w_in  = (const float*)d_in[2];
    const float* b_in  = (const float*)d_in[3];
    const float* Ws    = (const float*)d_in[4];
    const float* bs    = (const float*)d_in[5];
    const float* w_out = (const float*)d_in[6];
    const float* b_out = (const float*)d_in[7];
    float* y = (float*)d_out;

    _Float16* wq = (_Float16*)d_ws;            // 4*256*256*2 = 512 KB scratch
    int BT = in_sizes[0] / (NJ * 3);           // 64*243 = 15552

    int nw = LAYERS * H * H;
    pack_w_f16<<<(nw + 255) / 256, 256, 0, stream>>>(Ws, wq);

    int nblocks = (BT + NB - 1) / NB;          // 5184 workgroups
    gcn_fused<<<nblocks, 256, 0, stream>>>(x, adj, w_in, b_in, wq, bs, w_out, b_out, y, BT);
}
